// DualAttentionSSMKeys_66073776881797
// MI455X (gfx1250) — matrix operand order repaired and checked by reading, not compiled, not run on hardware
//
#include <hip/hip_runtime.h>

// ---------------------------------------------------------------- types ----
typedef __bf16 bf16_t;
typedef __attribute__((ext_vector_type(16))) bf16_t v16bf;
typedef __attribute__((ext_vector_type(8)))  bf16_t v8bf;
typedef __attribute__((ext_vector_type(8)))  float  v8f;

#define NB 2
#define NP 196
#define NF 8
#define ND 384
#define NT 1568          // F*P
#define NH 8
#define NDH 64
#define NINNER 512       // H*DH
#define NS 192
#define NR 16
#define NRS2 400         // R + 2S
#define NROWS (NB*NT)    // 3136
#define SCALE_QK 0.125f  // DH^-0.5
#define FSCALE 0.1f

static __device__ __forceinline__ unsigned short f2bf(float f) {
  union { float f; unsigned u; } v; v.f = f;
  unsigned u = v.u;
  u += 0x7FFFu + ((u >> 16) & 1u);   // round-to-nearest-even
  return (unsigned short)(u >> 16);
}

static __device__ __forceinline__ v8f zero8() {
  v8f z = {0.f, 0.f, 0.f, 0.f, 0.f, 0.f, 0.f, 0.f};
  return z;
}
static __device__ __forceinline__ v8f wmma_bf16(v16bf a, v16bf b, v8f c) {
  return __builtin_amdgcn_wmma_f32_16x16x32_bf16(false, a, false, b, (short)0, c,
                                                 false, false);
}
// A fragment (16x32 bf16): lanes 0-15 hold K 0-7 & 16-23, lanes 16-31 hold 8-15 & 24-31
static __device__ __forceinline__ v16bf ldfragA(const unsigned short* p) {
  v8bf lo = *reinterpret_cast<const v8bf*>(p);
  v8bf hi = *reinterpret_cast<const v8bf*>(p + 16);
  return __builtin_shufflevector(lo, hi, 0,1,2,3,4,5,6,7,8,9,10,11,12,13,14,15);
}
// B fragment (32x16 bf16): lanes 0-15 hold K 0-7 & 16-23, lanes 16-31 hold 8-15 & 24-31
static __device__ __forceinline__ v16bf ldfragB(const unsigned short* p) {
  v8bf lo = *reinterpret_cast<const v8bf*>(p);
  v8bf hi = *reinterpret_cast<const v8bf*>(p + 16);
  return __builtin_shufflevector(lo, hi, 0,1,2,3,4,5,6,7,8,9,10,11,12,13,14,15);
}
static __device__ __forceinline__ void store_tile(float* C, unsigned short* Cb, int ldc,
                                                  int m0, int n0, int l16, int half,
                                                  v8f acc, float alpha,
                                                  const float* bias) {
  const int n = n0 + l16;
  const float bv = bias ? bias[n] : 0.f;
  const size_t base = (size_t)(m0 + half * 8) * ldc + n;
  if (C) {
#pragma unroll
    for (int r = 0; r < 8; ++r) C[base + (size_t)r * ldc] = alpha * acc[r] + bv;
  }
  if (Cb) {
#pragma unroll
    for (int r = 0; r < 8; ++r) Cb[base + (size_t)r * ldc] = f2bf(alpha * acc[r] + bv);
  }
}

// --------------------------------- WMMA bf16 GEMM, 32x32 C-tile per wave ----
// C[M,N] = alpha * A[M,K] @ Bt[N,K]^T + bias[N];  M,N % 32 == 0, K % 32 == 0.
// grid = (N/32, M/32), block = 32 (one wave). 4 WMMA per 8 b128 loads.
__global__ __launch_bounds__(32)
void wmma_gemm2x2_kernel(const unsigned short* __restrict__ A, int lda,
                         const unsigned short* __restrict__ Bt, int ldb,
                         float* __restrict__ C, unsigned short* __restrict__ Cb,
                         int ldc, const float* __restrict__ bias,
                         int K, float alpha) {
  const int lane = threadIdx.x;
  const int half = lane >> 4;
  const int l16  = lane & 15;
  const int m0 = blockIdx.y * 32, n0 = blockIdx.x * 32;
  const unsigned short* a0p = A  + (size_t)(m0 + l16) * lda + half * 8;
  const unsigned short* a1p = a0p + (size_t)16 * lda;
  const unsigned short* b0p = Bt + (size_t)(n0 + l16) * ldb + half * 8;
  const unsigned short* b1p = b0p + (size_t)16 * ldb;
  v8f acc00 = zero8(), acc01 = zero8(), acc10 = zero8(), acc11 = zero8();
  for (int k = 0; k < K; k += 32) {
    v16bf a0 = ldfragA(a0p + k);
    v16bf a1 = ldfragA(a1p + k);
    v16bf b0 = ldfragB(b0p + k);
    v16bf b1 = ldfragB(b1p + k);
    acc00 = wmma_bf16(a0, b0, acc00);
    acc01 = wmma_bf16(a0, b1, acc01);
    acc10 = wmma_bf16(a1, b0, acc10);
    acc11 = wmma_bf16(a1, b1, acc11);
  }
  store_tile(C, Cb, ldc, m0,      n0,      l16, half, acc00, alpha, bias);
  store_tile(C, Cb, ldc, m0,      n0 + 16, l16, half, acc01, alpha, bias);
  store_tile(C, Cb, ldc, m0 + 16, n0,      l16, half, acc10, alpha, bias);
  store_tile(C, Cb, ldc, m0 + 16, n0 + 16, l16, half, acc11, alpha, bias);
}

// ---------------------- generic 16x16-per-wave GEMM (for N=400 proj case) ----
__global__ __launch_bounds__(32)
void wmma_gemm_kernel(const unsigned short* __restrict__ A, int lda,
                      const unsigned short* __restrict__ Bt, int ldb,
                      float* __restrict__ C, int ldc,
                      const float* __restrict__ bias,
                      int K, float alpha) {
  const int lane = threadIdx.x;
  const int half = lane >> 4;
  const int l16  = lane & 15;
  const unsigned short* arow = A  + (size_t)(blockIdx.y * 16 + l16) * lda + half * 8;
  const unsigned short* brow = Bt + (size_t)(blockIdx.x * 16 + l16) * ldb + half * 8;
  v8f acc = zero8();
  for (int k = 0; k < K; k += 32) {
    acc = wmma_bf16(ldfragA(arow + k), ldfragB(brow + k), acc);
  }
  const int n  = blockIdx.x * 16 + l16;
  const float bv = bias ? bias[n] : 0.f;
  float* cp = C + (size_t)(blockIdx.y * 16 + half * 8) * ldc + n;
#pragma unroll
  for (int r = 0; r < 8; ++r) cp[(size_t)r * ldc] = alpha * acc[r] + bv;
}

// ------------------------------------------------------------ LayerNorm ----
__global__ void ln_kernel(const float* __restrict__ x, const float* __restrict__ g,
                          const float* __restrict__ b, float* __restrict__ of,
                          unsigned short* __restrict__ ob) {
  __shared__ float rs[128], rq[128];
  const int row = blockIdx.x, tid = threadIdx.x;
  const float* xr = x + (size_t)row * ND;
  float s = 0.f, q = 0.f;
  for (int i = tid; i < ND; i += 128) { float v = xr[i]; s += v; q += v * v; }
  rs[tid] = s; rq[tid] = q; __syncthreads();
  for (int o = 64; o > 0; o >>= 1) {
    if (tid < o) { rs[tid] += rs[tid + o]; rq[tid] += rq[tid + o]; }
    __syncthreads();
  }
  const float mean = rs[0] / ND;
  const float var  = rq[0] / ND - mean * mean;
  const float inv  = rsqrtf(var + 1e-5f);
  for (int i = tid; i < ND; i += 128) {
    float v = (xr[i] - mean) * inv * g[i] + b[i];
    if (of) of[(size_t)row * ND + i] = v;
    ob[(size_t)row * ND + i] = f2bf(v);
  }
}

// ---------------------------------------------------- dt = softplus proj ----
__global__ void dt_kernel(const float* __restrict__ proj, const float* __restrict__ w,
                          const float* __restrict__ bias, float* __restrict__ dt) {
  const int i = blockIdx.x * blockDim.x + threadIdx.x;
  if (i >= NROWS * ND) return;
  const int r = i / ND, d = i % ND;
  float acc = bias[d];
#pragma unroll
  for (int j = 0; j < NR; ++j) acc += proj[(size_t)r * NRS2 + j] * w[(size_t)j * ND + d];
  dt[i] = (acc > 20.f) ? acc : log1pf(expf(acc));   // * STEP(=1.0)
}

// -------------------------------------------- Mamba scan, state in LDS -----
// grid = B*P blocks, block = D(=384) threads; 384*193*4 + 384*4 = ~298KB LDS.
// B_t/C_t staged each step with async global->LDS DMA (ASYNCcnt path).
__global__ void mamba_kernel(const float* __restrict__ xb, const float* __restrict__ proj,
                             const float* __restrict__ dt, const float* __restrict__ A_log,
                             const float* __restrict__ Dp, float* __restrict__ y1) {
  extern __shared__ float lds[];
  const int STRIDE = NS + 1;                 // 193: conflict-free (193 % 64 == 1)
  float* bc = lds + ND * STRIDE;             // [0,NS) = B_t, [NS,2*NS) = C_t (contiguous)
  float* cc = bc + NS;
  const int d = threadIdx.x;                 // 0..383  (== 2*NS)
  const int b = blockIdx.x / NP, p = blockIdx.x % NP;
  float* hd = lds + d * STRIDE;
  for (int s = 0; s < NS; ++s) hd[s] = 0.f;
  const float* Ar = A_log + (size_t)d * NS;
  const float dpd = Dp[d];
  const unsigned bc_lds = (unsigned)(unsigned long long)(bc + d);
  for (int f = 0; f < NF; ++f) {
    const int row = (b * NF + f) * NP + p;
    // LDS[bc+d] = proj[row*400 + 16 + d] for d in [0,384): covers B_t then C_t.
    {
      const unsigned long long gsrc =
          (unsigned long long)(proj + (size_t)row * NRS2 + NR + d);
      asm volatile("global_load_async_to_lds_b32 %0, %1, off"
                   :: "v"(bc_lds), "v"(gsrc) : "memory");
      asm volatile("s_wait_asynccnt 0" ::: "memory");
    }
    __syncthreads();
    const float dtv = dt[(size_t)row * ND + d];
    const float xv  = xb[(size_t)row * ND + d];
    const float dx  = dtv * xv;
    float y = 0.f;
    for (int s = 0; s < NS; ++s) {
      const float a = -expf(Ar[s]);
      const float h = expf(dtv * a) * hd[s] + dx * bc[s];
      hd[s] = h;
      y += h * cc[s];
    }
    y1[(size_t)row * ND + d] = y + dpd * xv + xv;
    __syncthreads();
  }
}

// ---------------------------------------- softmax(sb) + 0.1*softmax(sa) ----
__global__ void softmax_combine_kernel(const float* __restrict__ sb,
                                       const float* __restrict__ sa,
                                       unsigned short* __restrict__ attn) {
  __shared__ float r1[256], r2[256];
  const int n = blockIdx.x, tid = threadIdx.x;
  const int L = (n / NP + 1) * NP;           // block-causal prefix length
  const float* rb = sb + (size_t)n * NT;
  const float* ra = sa + (size_t)n * NT;
  float mb = -3.0e38f, ma = -3.0e38f;
  for (int i = tid; i < L; i += 256) { mb = fmaxf(mb, rb[i]); ma = fmaxf(ma, ra[i]); }
  r1[tid] = mb; r2[tid] = ma; __syncthreads();
  for (int o = 128; o > 0; o >>= 1) {
    if (tid < o) { r1[tid] = fmaxf(r1[tid], r1[tid + o]); r2[tid] = fmaxf(r2[tid], r2[tid + o]); }
    __syncthreads();
  }
  mb = r1[0]; ma = r2[0]; __syncthreads();
  float s1 = 0.f, s2 = 0.f;
  for (int i = tid; i < L; i += 256) { s1 += expf(rb[i] - mb); s2 += expf(ra[i] - ma); }
  r1[tid] = s1; r2[tid] = s2; __syncthreads();
  for (int o = 128; o > 0; o >>= 1) {
    if (tid < o) { r1[tid] += r1[tid + o]; r2[tid] += r2[tid + o]; }
    __syncthreads();
  }
  const float ib = 1.f / r1[0], ia = FSCALE / r2[0];
  unsigned short* orow = attn + (size_t)n * NT;
  for (int i = tid; i < NT; i += 256) {
    float v = 0.f;
    if (i < L) v = expf(rb[i] - mb) * ib + expf(ra[i] - ma) * ia;
    orow[i] = f2bf(v);
  }
}

// ----------------------------------------------------- elementwise utils ----
__global__ void gelu_f2bf_kernel(const float* __restrict__ s, unsigned short* __restrict__ d, int n) {
  int i = blockIdx.x * blockDim.x + threadIdx.x;
  if (i < n) { float v = s[i]; d[i] = f2bf(0.5f * v * (1.f + erff(v * 0.70710678118654752f))); }
}
__global__ void add_convert_kernel(const float* __restrict__ a, const float* __restrict__ b,
                                   float* __restrict__ of, unsigned short* __restrict__ ob, int n) {
  int i = blockIdx.x * blockDim.x + threadIdx.x;
  if (i < n) { float v = a[i] + b[i]; of[i] = v; ob[i] = f2bf(v); }
}
// dst[n*K+k] = bf16(src[k*N+n])  (weight W[K,N] -> Wt[N,K])
__global__ void transpose_w_kernel(const float* __restrict__ src, unsigned short* __restrict__ dst,
                                   int K, int N) {
  int i = blockIdx.x * blockDim.x + threadIdx.x;
  if (i >= K * N) return;
  int n = i % N, k = i / N;
  dst[(size_t)n * K + k] = f2bf(src[i]);
}
// Vt[b,h,dh,t] = Vb16[b*T+t, h*DH+dh]  (bf16 bits pass-through)
__global__ void vtrans_kernel(const unsigned short* __restrict__ V, unsigned short* __restrict__ Vt) {
  int i = blockIdx.x * blockDim.x + threadIdx.x;
  const int total = NB * NH * NDH * NT;
  if (i >= total) return;
  int t = i % NT; int r = i / NT;
  int dh = r % NDH; r /= NDH;
  int h = r % NH;  int b = r / NH;
  Vt[i] = V[((size_t)b * NT + t) * NINNER + h * NDH + dh];
}

// ------------------------------------------------------------------ host ----
static inline size_t alignup(size_t x) { return (x + 255) & ~(size_t)255; }

extern "C" void kernel_launch(void* const* d_in, const int* in_sizes, int n_in,
                              void* d_out, int out_size, void* d_ws, size_t ws_size,
                              hipStream_t stream) {
  (void)in_sizes; (void)n_in; (void)out_size; (void)ws_size;
  const float* x        = (const float*)d_in[0];
  const float* norm_g   = (const float*)d_in[1];
  const float* norm_b   = (const float*)d_in[2];
  const float* Wq       = (const float*)d_in[3];
  const float* Wk_cnt   = (const float*)d_in[4];
  const float* Wv       = (const float*)d_in[5];
  const float* Wk_mem   = (const float*)d_in[6];
  const float* Wo       = (const float*)d_in[7];
  const float* bo       = (const float*)d_in[8];
  const float* xproj_w  = (const float*)d_in[9];
  const float* xproj_b  = (const float*)d_in[10];
  const float* dtproj_w = (const float*)d_in[11];
  const float* dtproj_b = (const float*)d_in[12];
  const float* A_log    = (const float*)d_in[13];
  const float* Dp       = (const float*)d_in[14];
  const float* ln2_g    = (const float*)d_in[15];
  const float* ln2_b    = (const float*)d_in[16];
  const float* W1       = (const float*)d_in[17];
  const float* b1       = (const float*)d_in[18];
  const float* W2       = (const float*)d_in[19];
  const float* b2       = (const float*)d_in[20];
  // d_in[21] = mask (computed analytically)

  char* ws = (char*)d_ws;
  size_t cur = 0;
#define ALLOC(ptr_t, name, nelem, esz) \
  ptr_t name = (ptr_t)(ws + cur); cur = alignup(cur + (size_t)(nelem) * (esz));
  ALLOC(float*,          xbf,   (size_t)NROWS * ND, 4)
  ALLOC(unsigned short*, xbb,   (size_t)NROWS * ND, 2)
  ALLOC(unsigned short*, WqT,   (size_t)ND * NINNER, 2)
  ALLOC(unsigned short*, WkcT,  (size_t)ND * NINNER, 2)
  ALLOC(unsigned short*, WvT,   (size_t)ND * NINNER, 2)
  ALLOC(unsigned short*, WkmT,  (size_t)ND * NINNER, 2)
  ALLOC(unsigned short*, XprT,  (size_t)ND * NRS2, 2)
  ALLOC(unsigned short*, W1T,   (size_t)ND * 2 * ND, 2)
  ALLOC(unsigned short*, W2T,   (size_t)2 * ND * ND, 2)
  ALLOC(unsigned short*, WoT,   (size_t)NINNER * ND, 2)
  ALLOC(unsigned short*, Qb,    (size_t)NROWS * NINNER, 2)
  ALLOC(unsigned short*, Kcb,   (size_t)NROWS * NINNER, 2)
  ALLOC(unsigned short*, Vb,    (size_t)NROWS * NINNER, 2)
  ALLOC(unsigned short*, Vtb,   (size_t)NROWS * NINNER, 2)
  ALLOC(float*,          projf, (size_t)NROWS * NRS2, 4)
  ALLOC(float*,          dtf,   (size_t)NROWS * ND, 4)
  ALLOC(float*,          y1f,   (size_t)NROWS * ND, 4)
  ALLOC(unsigned short*, h2b,   (size_t)NROWS * ND, 2)
  ALLOC(float*,          midf,  (size_t)NROWS * 2 * ND, 4)
  ALLOC(unsigned short*, midb,  (size_t)NROWS * 2 * ND, 2)
  ALLOC(float*,          mlpf,  (size_t)NROWS * ND, 4)
  ALLOC(float*,          y2f,   (size_t)NROWS * ND, 4)
  ALLOC(unsigned short*, y2b,   (size_t)NROWS * ND, 2)
  ALLOC(unsigned short*, Kmb,   (size_t)NROWS * NINNER, 2)
  ALLOC(float*,          sbf,   (size_t)NT * NT, 4)
  ALLOC(float*,          saf,   (size_t)NT * NT, 4)
  ALLOC(unsigned short*, attnb, (size_t)NT * NT, 2)
  ALLOC(unsigned short*, Outb,  (size_t)NROWS * NINNER, 2)
#undef ALLOC

  const dim3 b32(32), b128(128), b256(256);
#define EWG(n) dim3(((n) + 255) / 256)
#define G2(Nn, Mm) dim3((Nn) / 32, (Mm) / 32)

  // 1) weights -> transposed bf16
  transpose_w_kernel<<<EWG(ND * NINNER), b256, 0, stream>>>(Wq,      WqT,  ND, NINNER);
  transpose_w_kernel<<<EWG(ND * NINNER), b256, 0, stream>>>(Wk_cnt,  WkcT, ND, NINNER);
  transpose_w_kernel<<<EWG(ND * NINNER), b256, 0, stream>>>(Wv,      WvT,  ND, NINNER);
  transpose_w_kernel<<<EWG(ND * NINNER), b256, 0, stream>>>(Wk_mem,  WkmT, ND, NINNER);
  transpose_w_kernel<<<EWG(ND * NRS2),   b256, 0, stream>>>(xproj_w, XprT, ND, NRS2);
  transpose_w_kernel<<<EWG(ND * 2 * ND), b256, 0, stream>>>(W1,      W1T,  ND, 2 * ND);
  transpose_w_kernel<<<EWG(2 * ND * ND), b256, 0, stream>>>(W2,      W2T,  2 * ND, ND);
  transpose_w_kernel<<<EWG(NINNER * ND), b256, 0, stream>>>(Wo,      WoT,  NINNER, ND);

  // 2) LN1
  ln_kernel<<<NROWS, b128, 0, stream>>>(x, norm_g, norm_b, xbf, xbb);

  // 3) Q / Kc / V (bf16 out, fused) and proj (f32 + bias, N=400 path)
  wmma_gemm2x2_kernel<<<G2(NINNER, NROWS), b32, 0, stream>>>(
      xbb, ND, WqT, ND, nullptr, Qb, NINNER, nullptr, ND, 1.f);
  wmma_gemm2x2_kernel<<<G2(NINNER, NROWS), b32, 0, stream>>>(
      xbb, ND, WkcT, ND, nullptr, Kcb, NINNER, nullptr, ND, 1.f);
  wmma_gemm2x2_kernel<<<G2(NINNER, NROWS), b32, 0, stream>>>(
      xbb, ND, WvT, ND, nullptr, Vb, NINNER, nullptr, ND, 1.f);
  wmma_gemm_kernel<<<dim3(NRS2 / 16, NROWS / 16), b32, 0, stream>>>(
      xbb, ND, XprT, ND, projf, NRS2, xproj_b, ND, 1.f);
  vtrans_kernel<<<EWG(NROWS * NINNER), b256, 0, stream>>>(Vb, Vtb);

  // 4) dt + Mamba scan (state in 298KB of CDNA5 LDS, async B/C staging)
  dt_kernel<<<EWG(NROWS * ND), b256, 0, stream>>>(projf, dtproj_w, dtproj_b, dtf);
  const size_t mambaLds = ((size_t)ND * (NS + 1) + 2 * NS) * sizeof(float);
  mamba_kernel<<<NB * NP, dim3(ND), mambaLds, stream>>>(xbf, projf, dtf, A_log, Dp, y1f);

  // 5) LN2 + MLP + residual
  ln_kernel<<<NROWS, b128, 0, stream>>>(y1f, ln2_g, ln2_b, nullptr, h2b);
  wmma_gemm2x2_kernel<<<G2(2 * ND, NROWS), b32, 0, stream>>>(
      h2b, ND, W1T, ND, midf, nullptr, 2 * ND, b1, ND, 1.f);
  gelu_f2bf_kernel<<<EWG(NROWS * 2 * ND), b256, 0, stream>>>(midf, midb, NROWS * 2 * ND);
  wmma_gemm2x2_kernel<<<G2(ND, NROWS), b32, 0, stream>>>(
      midb, 2 * ND, W2T, 2 * ND, mlpf, nullptr, ND, b2, 2 * ND, 1.f);
  add_convert_kernel<<<EWG(NROWS * ND), b256, 0, stream>>>(mlpf, y1f, y2f, y2b, NROWS * ND);

  // 6) memory keys (bf16 out, fused)
  wmma_gemm2x2_kernel<<<G2(NINNER, NROWS), b32, 0, stream>>>(
      y2b, ND, WkmT, ND, nullptr, Kmb, NINNER, nullptr, ND, 1.f);

  // 7) dual attention, per (b,h); combined probs -> single PV GEMM (bf16 out)
  for (int b = 0; b < NB; ++b) {
    for (int h = 0; h < NH; ++h) {
      const unsigned short* Qh  = Qb  + (size_t)b * NT * NINNER + h * NDH;
      const unsigned short* Kch = Kcb + (size_t)b * NT * NINNER + h * NDH;
      const unsigned short* Kmh = Kmb + (size_t)b * NT * NINNER + h * NDH;
      const unsigned short* Vth = Vtb + ((size_t)b * NH + h) * NDH * NT;
      wmma_gemm2x2_kernel<<<G2(NT, NT), b32, 0, stream>>>(
          Qh, NINNER, Kch, NINNER, sbf, nullptr, NT, nullptr, NDH, SCALE_QK);
      wmma_gemm2x2_kernel<<<G2(NT, NT), b32, 0, stream>>>(
          Qh, NINNER, Kmh, NINNER, saf, nullptr, NT, nullptr, NDH, SCALE_QK);
      softmax_combine_kernel<<<NT, b256, 0, stream>>>(sbf, saf, attnb);
      wmma_gemm2x2_kernel<<<G2(NDH, NT), b32, 0, stream>>>(
          attnb, NT, Vth, NT, nullptr, Outb + (size_t)b * NT * NINNER + h * NDH,
          NINNER, nullptr, NT, 1.f);
    }
  }

  // 8) output projection -> d_out (fp32)
  wmma_gemm2x2_kernel<<<G2(ND, NROWS), b32, 0, stream>>>(
      Outb, NINNER, WoT, NINNER, (float*)d_out, nullptr, ND, bo, NINNER, 1.f);
#undef EWG
#undef G2
}
